// DQN_21251498181475
// MI455X (gfx1250) — matrix operand order repaired and checked by reading, not compiled, not run on hardware
//
#include <hip/hip_runtime.h>
#include <hip/hip_bf16.h>

// ---------------------------------------------------------------------------
// DQN LSTM+MLP for MI455X (gfx1250, wave32, WMMA)
//
// Persistent-grid LSTM, W_hh resident in LDS (8 MB f16 across 32 WGPs =
// 256 KB each < 320 KB CDNA5 LDS). Per-step matvec via
// v_wmma_f32_16x16x32_f16 with h in the A operand (replicated across M rows:
// one lane-uniform 32B LDS load, zero splat moves) and weights in the B
// operand (pre-packed per-lane planes). Per-step h staging and weight
// prefill use CDNA5 async global->LDS loads.
// ---------------------------------------------------------------------------

typedef __attribute__((ext_vector_type(16))) _Float16 v16h;
typedef __attribute__((ext_vector_type(8)))  _Float16 v8h;
typedef __attribute__((ext_vector_type(8)))  float    v8f;

#define HSZ   1000   // hidden size
#define DSZ   18     // input feature size
#define HP    1024   // padded hidden
#define NWG   32     // persistent workgroups
#define UPW   32     // units per workgroup  (NWG*UPW == HP)
#define RPW   128    // gate rows per workgroup (4*UPW)
#define RT    8      // 16-row WMMA tiles per workgroup (RPW/16)
#define KT    32     // 32-K WMMA tiles (HP/32)

// workspace byte offsets
#define WPK_OFF   0u                          // packed W_hh f16 (B-operand layout)
#define WPK_BYTES (NWG * RT * KT * 1024u)     // 8,388,608
#define HBUF_OFF  (WPK_OFF + WPK_BYTES)       // double-buffered h, f16[2][1024]
#define HFIN_OFF  (HBUF_OFF + 4096u)          // final h fp32[1024]
#define BAR_OFF   (HFIN_OFF + 4096u)          // barrier counter (u32)

// LDS byte offsets inside the persistent kernel
#define L_W     0u            // packed weights: 262144 B
#define L_H     262144u       // permuted h stage: f16[1024] = 2048 B
#define L_XG    264192u       // xg: f32[128]
#define L_GAT   264704u       // wmma gates: f32[128]
#define L_XT    265216u       // x_t stage: f32[32]
#define L_WIH   265344u       // W_ih slice: f32[128*18] = 9216 B
#define L_BIA   274560u       // fused bias: f32[128]
#define L_TOTAL 275072u       // < 320 KB

// ---------------------------------------------------------------------------
// CDNA5 async global->LDS (guarded; falls back to plain copies)
// ---------------------------------------------------------------------------
#if defined(__has_builtin)
#if __has_builtin(__builtin_amdgcn_global_load_async_to_lds_b32) && \
    __has_builtin(__builtin_amdgcn_global_load_async_to_lds_b128) && \
    __has_builtin(__builtin_amdgcn_s_wait_asynccnt)
#define USE_ASYNC_LDS 1
#endif
#endif

// native int vector matching the builtin's expected type:
// "__attribute__((__vector_size__(4 * sizeof(int)))) int"
typedef int nvi4 __attribute__((vector_size(16)));

#ifdef USE_ASYNC_LDS
typedef __attribute__((address_space(1))) int  as1_i32;
typedef __attribute__((address_space(3))) int  as3_i32;
typedef __attribute__((address_space(1))) nvi4 as1_v4i;
typedef __attribute__((address_space(3))) nvi4 as3_v4i;
#endif

__device__ __forceinline__ void copy_b32_to_lds(const unsigned* g, unsigned* l) {
#ifdef USE_ASYNC_LDS
    __builtin_amdgcn_global_load_async_to_lds_b32((as1_i32*)g, (as3_i32*)l, 0, 0);
#else
    *l = *g;
#endif
}
__device__ __forceinline__ void copy_b128_to_lds(const nvi4* g, nvi4* l) {
#ifdef USE_ASYNC_LDS
    __builtin_amdgcn_global_load_async_to_lds_b128((as1_v4i*)g, (as3_v4i*)l, 0, 0);
#else
    *l = *g;
#endif
}
__device__ __forceinline__ void wait_lds_copies() {
#ifdef USE_ASYNC_LDS
    __builtin_amdgcn_s_wait_asynccnt(0);
#endif
}

// ---------------------------------------------------------------------------
// Pack W_hh (fp32 [4H,H], gate order i,f,g,o) into the CDNA5 16-bit B-matrix
// 32x16 layout: elements = K within tile, lane%16 = output column n (gate row
// rt*16+n). Each 1024B block (rowtile, ktile) is split into two 512B planes
// (K 8h+0..7, K 16+8h+0..7, h = lane/16), 16B contiguous per lane.
// Rows are gate-major within a WG: row = gate*32 + unit_local.
// ---------------------------------------------------------------------------
__global__ void pack_whh_kernel(const float* __restrict__ Whh,
                                _Float16* __restrict__ wpk) {
    int tid  = blockIdx.x * blockDim.x + threadIdx.x;   // 262144 lane-tasks
    int lane = tid & 31;
    int blk  = tid >> 5;                                // (wg*8+rt)*32+kt
    int kt   = blk & 31;
    int rt   = (blk >> 5) & 7;
    int wg   = blk >> 8;
    int kb   = kt * 32 + 8 * (lane >> 4);               // lane%16 == column, K half 8*(lane/16)

    v8h lo, hi;
    int row = rt * 16 + (lane & 15);
    int g = row >> 5, u = wg * UPW + (row & 31);
#pragma unroll
    for (int n = 0; n < 8; ++n) {
        int k0 = kb + n;
        int k1 = k0 + 16;
        float w0 = 0.0f, w1 = 0.0f;
        if (u < HSZ) {
            if (k0 < HSZ) w0 = Whh[(size_t)(g * HSZ + u) * HSZ + k0];
            if (k1 < HSZ) w1 = Whh[(size_t)(g * HSZ + u) * HSZ + k1];
        }
        lo[n] = (_Float16)w0;
        hi[n] = (_Float16)w1;
    }
    _Float16* base = wpk + (size_t)blk * 512;
    *(v8h*)(base + lane * 8)       = lo;   // plane 0: K 8h+0..7
    *(v8h*)(base + 256 + lane * 8) = hi;   // plane 1: K 16+8h+0..7
}

// ---------------------------------------------------------------------------
// Init h double buffer (f16, zero padded) and barrier counter.
// ---------------------------------------------------------------------------
__global__ void lstm_init_kernel(const float* __restrict__ h0,
                                 _Float16* __restrict__ hbuf,
                                 float* __restrict__ hfin,
                                 unsigned* __restrict__ bar) {
    int j = blockIdx.x * blockDim.x + threadIdx.x;      // 1024 threads
    if (j < HP) {
        hbuf[j]      = (j < HSZ) ? (_Float16)h0[j] : (_Float16)0.0f;
        hbuf[HP + j] = (_Float16)0.0f;
        hfin[j]      = 0.0f;
    }
    if (j == 0) *bar = 0u;
}

// ---------------------------------------------------------------------------
// Persistent LSTM scan. 32 blocks x 256 threads (8 wave32s).
// h is staged into LDS pre-permuted so the A fragment (h replicated across
// the 16 M rows) is one contiguous lane-uniform 32B load per K tile:
// per 32-K tile, u32 quads reordered [0,2,1,3] => halves [0..7,16..23 |
// 8..15,24..31], matching the ISA 16-bit A-matrix K striping.
// ---------------------------------------------------------------------------
__launch_bounds__(256, 1)
__global__ void lstm_persistent_kernel(const float* __restrict__ x,
                                       const float* __restrict__ c0,
                                       const float* __restrict__ Wih,
                                       const float* __restrict__ bih,
                                       const float* __restrict__ bhh,
                                       const _Float16* __restrict__ wpk,
                                       _Float16* __restrict__ hbuf,
                                       float* __restrict__ hfin,
                                       unsigned* __restrict__ bar,
                                       int T) {
    extern __shared__ char smem[];
    _Float16* wlds  = (_Float16*)(smem + L_W);
    _Float16* hlds  = (_Float16*)(smem + L_H);
    float*    xg    = (float*)(smem + L_XG);
    float*    gat   = (float*)(smem + L_GAT);
    float*    xt    = (float*)(smem + L_XT);
    float*    wih_l = (float*)(smem + L_WIH);
    float*    bia_l = (float*)(smem + L_BIA);

    const int tid  = threadIdx.x;
    const int wg   = blockIdx.x;
    const int lane = tid & 31;
    const int wv   = tid >> 5;       // wave id == row tile id

    // ---- one-time prefill: weight slice (256 KB) into LDS (async) ----
    {
        const nvi4* src = (const nvi4*)(wpk + (size_t)wg * (RT * KT * 512));
        nvi4*       dst = (nvi4*)wlds;
        for (int i = tid; i < (RT * KT * 1024) / 16; i += 256)
            copy_b128_to_lds(src + i, dst + i);
    }
    // W_ih slice + fused bias into LDS
    for (int i = tid; i < RPW * DSZ; i += 256) {
        int row = i / DSZ, d = i - row * DSZ;
        int gate = row >> 5, unit = wg * UPW + (row & 31);
        wih_l[i] = (unit < HSZ) ? Wih[(size_t)(gate * HSZ + unit) * DSZ + d] : 0.0f;
    }
    if (tid < RPW) {
        int gate = tid >> 5, unit = wg * UPW + (tid & 31);
        bia_l[tid] = (unit < HSZ) ? (bih[gate * HSZ + unit] + bhh[gate * HSZ + unit]) : 0.0f;
    }

    const int unit = wg * UPW + (tid & 31);   // meaningful for tid < 32
    float c = 0.0f;
    if (tid < UPW && unit < HSZ) c = c0[unit];
    wait_lds_copies();
    __syncthreads();

    for (int s = 0; s < T; ++s) {
        // ---- grid-wide sync: all WGs finished step s-1 ----
        if (s > 0) {
            if (tid == 0) {
                unsigned target = (unsigned)NWG * (unsigned)s;
                while (__hip_atomic_load(bar, __ATOMIC_ACQUIRE,
                                         __HIP_MEMORY_SCOPE_AGENT) < target)
                    __builtin_amdgcn_s_sleep(1);
            }
            __syncthreads();
            __threadfence();   // acquire published h across agents
        }

        // ---- stage h (permuted) and x_t into LDS (async b32) ----
        {
            const unsigned* hsrc = (const unsigned*)(hbuf + (size_t)(s & 1) * HP);
            unsigned*       hdst = (unsigned*)hlds;
            for (int i = tid; i < HP / 2; i += 256) {
                int j = i & 15;
                int g = j >> 2;                       // u32 quad within tile
                int tg = ((g & 1) << 1) | (g >> 1);   // [0,2,1,3] reorder
                copy_b32_to_lds(hsrc + i, hdst + ((i & ~15) | (tg << 2) | (j & 3)));
            }
        }
        if (tid < DSZ)
            copy_b32_to_lds((const unsigned*)(x + (size_t)s * DSZ + tid),
                            (unsigned*)(xt + tid));
        wait_lds_copies();
        __syncthreads();

        // ---- input projection xg[row] = b + x_t . W_ih[row] ----
        if (tid < RPW) {
            float a = bia_l[tid];
            const float* wr = wih_l + tid * DSZ;
#pragma unroll
            for (int d = 0; d < DSZ; ++d) a += wr[d] * xt[d];
            xg[tid] = a;
        }

        // ---- WMMA matvec: row tile wv, accumulate over 32 K tiles ----
        // A = h replicated across M (lane-uniform 32B load per half-wave),
        // B = weight tile (two 16B planes per lane).
        v8f acc = {};
        const char* hbase = (const char*)hlds + ((lane >> 4) << 5);
        const char* wbase = (const char*)wlds + ((size_t)wv << 15);  // wv*KT*1024
#pragma unroll 4
        for (int t = 0; t < KT; ++t) {
            v16h a = *(const v16h*)(hbase + t * 64);
            v8h blo = *(const v8h*)(wbase + (t << 10) + lane * 16);
            v8h bhi = *(const v8h*)(wbase + (t << 10) + 512 + lane * 16);
            v16h b = __builtin_shufflevector(blo, bhi,
                0, 1, 2, 3, 4, 5, 6, 7, 8, 9, 10, 11, 12, 13, 14, 15);
            acc = __builtin_amdgcn_wmma_f32_16x16x32_f16(
                false, a, false, b, (short)0, acc, false, false);
        }
        // D[m,n] = gate[rowtile*16+n] replicated over m: lane n holds it.
        if (lane < 16) gat[wv * 16 + lane] = acc[0];
        __syncthreads();

        // ---- gate nonlinearities + state update (threads 0..31) ----
        if (tid < UPW) {
            float gi = gat[tid]            + xg[tid];
            float gf = gat[UPW + tid]      + xg[UPW + tid];
            float gg = gat[2 * UPW + tid]  + xg[2 * UPW + tid];
            float go = gat[3 * UPW + tid]  + xg[3 * UPW + tid];
            float i_ = 1.0f / (1.0f + __expf(-gi));
            float f_ = 1.0f / (1.0f + __expf(-gf));
            float g_ = tanhf(gg);
            float o_ = 1.0f / (1.0f + __expf(-go));
            c = f_ * c + i_ * g_;
            float h = o_ * tanhf(c);
            hbuf[(size_t)((s + 1) & 1) * HP + unit] = (_Float16)h;
            if (s == T - 1 && unit < HSZ) hfin[unit] = h;
        }

        // ---- publish + arrive ----
        __threadfence();
        __syncthreads();
        if (tid == 0)
            __hip_atomic_fetch_add(bar, 1u, __ATOMIC_RELEASE,
                                   __HIP_MEMORY_SCOPE_AGENT);
    }
}

// ---------------------------------------------------------------------------
// MLP head: relu(h) -> 4x relu(Linear) -> softmax(Linear[3]).  ~4 MFLOP.
// ---------------------------------------------------------------------------
__launch_bounds__(256, 1)
__global__ void mlp_head_kernel(const float* __restrict__ hfin,
                                const float* __restrict__ W1, const float* __restrict__ b1,
                                const float* __restrict__ W2, const float* __restrict__ b2,
                                const float* __restrict__ W3, const float* __restrict__ b3,
                                const float* __restrict__ W4, const float* __restrict__ b4,
                                const float* __restrict__ Wo, const float* __restrict__ bo,
                                float* __restrict__ out) {
    __shared__ float ta[HSZ];
    __shared__ float tb[HSZ];
    __shared__ float logits[3];
    const int tid = threadIdx.x;

    for (int j = tid; j < HSZ; j += 256) ta[j] = fmaxf(hfin[j], 0.0f);
    __syncthreads();

    const float* Ws[4] = {W1, W2, W3, W4};
    const float* bs[4] = {b1, b2, b3, b4};
    float* cur = ta;
    float* nxt = tb;
#pragma unroll
    for (int l = 0; l < 4; ++l) {
        for (int j = tid; j < HSZ; j += 256) {
            float a = bs[l][j];
            const float* wr = Ws[l] + (size_t)j * HSZ;
            for (int k = 0; k < HSZ; ++k) a += wr[k] * cur[k];
            nxt[j] = fmaxf(a, 0.0f);
        }
        __syncthreads();
        float* tmp = cur; cur = nxt; nxt = tmp;
    }

    if (tid < 3) {
        float a = bo[tid];
        const float* wr = Wo + (size_t)tid * HSZ;
        for (int k = 0; k < HSZ; ++k) a += wr[k] * cur[k];
        logits[tid] = a;
    }
    __syncthreads();
    if (tid == 0) {
        float m  = fmaxf(logits[0], fmaxf(logits[1], logits[2]));
        float e0 = __expf(logits[0] - m);
        float e1 = __expf(logits[1] - m);
        float e2 = __expf(logits[2] - m);
        float s  = e0 + e1 + e2;
        out[0] = e0 / s;
        out[1] = e1 / s;
        out[2] = e2 / s;
    }
}

// ---------------------------------------------------------------------------
extern "C" void kernel_launch(void* const* d_in, const int* in_sizes, int n_in,
                              void* d_out, int out_size, void* d_ws, size_t ws_size,
                              hipStream_t stream) {
    (void)n_in; (void)out_size; (void)ws_size;
    const float* x   = (const float*)d_in[0];
    const float* h0  = (const float*)d_in[1];
    const float* c0  = (const float*)d_in[2];
    const float* Wih = (const float*)d_in[3];
    const float* Whh = (const float*)d_in[4];
    const float* bih = (const float*)d_in[5];
    const float* bhh = (const float*)d_in[6];
    const float* W1  = (const float*)d_in[7];  const float* b1 = (const float*)d_in[8];
    const float* W2  = (const float*)d_in[9];  const float* b2 = (const float*)d_in[10];
    const float* W3  = (const float*)d_in[11]; const float* b3 = (const float*)d_in[12];
    const float* W4  = (const float*)d_in[13]; const float* b4 = (const float*)d_in[14];
    const float* Wo  = (const float*)d_in[15]; const float* bo = (const float*)d_in[16];
    float* out = (float*)d_out;

    const int T = in_sizes[0] / DSZ;   // 16384

    char* ws = (char*)d_ws;
    _Float16* wpk  = (_Float16*)(ws + WPK_OFF);
    _Float16* hbuf = (_Float16*)(ws + HBUF_OFF);
    float*    hfin = (float*)(ws + HFIN_OFF);
    unsigned* bar  = (unsigned*)(ws + BAR_OFF);

    pack_whh_kernel<<<1024, 256, 0, stream>>>(Whh, wpk);
    lstm_init_kernel<<<4, 256, 0, stream>>>(h0, hbuf, hfin, bar);
    lstm_persistent_kernel<<<NWG, 256, L_TOTAL, stream>>>(
        x, c0, Wih, bih, bhh, wpk, hbuf, hfin, bar, T);
    mlp_head_kernel<<<1, 256, 0, stream>>>(
        hfin, W1, b1, W2, b2, W3, b3, W4, b4, Wo, bo, out);
}
